// RingAttention_6992206758353
// MI455X (gfx1250) — matrix operand order repaired and checked by reading, not compiled, not run on hardware
//
#include <hip/hip_runtime.h>

typedef _Float16 HALF;
typedef __attribute__((ext_vector_type(16))) _Float16 v16h;
typedef __attribute__((ext_vector_type(8)))  float    v8f;
typedef __attribute__((ext_vector_type(4)))  unsigned int v4u;
typedef __attribute__((ext_vector_type(8)))  int          v8i;
typedef __attribute__((ext_vector_type(4)))  int          v4i;

// ---------------- WMMA helpers (CDNA5 wave32 layouts) ----------------

static __device__ __forceinline__ v8f wmma_f16(v16h a, v16h b, v8f c) {
  // D = A(16x32 f16) * B(32x16 f16) + C(16x16 f32)
  return __builtin_amdgcn_wmma_f32_16x16x32_f16(
      false, a, false, b, (short)0, c, false, false);
}

// A fragment (16x32): lane l holds row m = l&15.
// halves 0..15 -> K = (l<16 ? 0 : 16) + i
static __device__ __forceinline__ v16h gather_a(const HALF* __restrict__ base, int l) {
  const int hi = (l & 16) ? 1 : 0;
  const HALF* p0 = base + (hi ? 16 : 0);
  v16h a;
#pragma unroll
  for (int i = 0; i < 16; ++i) a[i] = p0[i];
  return a;
}

// B fragment (32x16): lane l holds col n = l&15; K = (l<16 ? 0 : 16) + i.
// base points at element [n][0], contiguous over the contraction dim.
static __device__ __forceinline__ v16h gather_b(const HALF* __restrict__ base, int l) {
  const HALF* p = base + ((l & 16) ? 16 : 0);
  v16h b;
#pragma unroll
  for (int i = 0; i < 16; ++i) b[i] = p[i];
  return b;
}

// B fragment from a row-major [K][N] tile: element [k][n] at base[k*stride],
// base already offset by n. Contraction index strided.
static __device__ __forceinline__ v16h gather_b_strided(const HALF* __restrict__ base,
                                                        int stride, int l) {
  const HALF* p = base + (((l & 16) ? 16 : 0) * stride);
  v16h b;
#pragma unroll
  for (int i = 0; i < 16; ++i) b[i] = p[i * stride];
  return b;
}

// ---------------- TDM: async 64x64 f16 tile, global -> LDS ----------------
// D# per CDNA5 ISA ch.8: group0 {count, lds_addr, global_addr, type=2},
// group1 {data_size=2B, tensor 64x64, tile 64x64, dim0_stride=64}, groups 2/3 zero (2-D).
static __device__ __forceinline__ void tdm_load_tile_64x64_f16(unsigned int lds_off,
                                                               const HALF* gptr) {
  const unsigned long long ga = (unsigned long long)(size_t)gptr;
  v4u g0;
  g0[0] = 1u;                                            // count=1, user descriptor
  g0[1] = lds_off;                                       // lds_addr (bytes)
  g0[2] = (unsigned int)(ga & 0xFFFFFFFFu);              // global_addr[31:0]
  g0[3] = (unsigned int)((ga >> 32) & 0x1FFFFFFu)        // global_addr[56:32]
          | (2u << 30);                                  // type=2 ("image")
  v8i g1;
  g1[0] = 0x10000;            // workgroup_mask=0, data_size=1 (2 bytes)
  g1[1] = (int)(64u << 16);   // tensor_dim0 = 64  (bits 79:48, low half)
  g1[2] = (int)(64u << 16);   // tensor_dim1 = 64  (bits 111:80, low half)
  g1[3] = (int)(64u << 16);   // tile_dim0   = 64  (bits 127:112)
  g1[4] = 64;                 // tile_dim1   = 64  (bits 143:128)
  g1[5] = 64;                 // tensor_dim0_stride = 64 (bits 207:160, low half)
  g1[6] = 0;
  g1[7] = 0;
  v4i z4 = {0, 0, 0, 0};
#if defined(__clang_major__) && (__clang_major__ >= 23)
  v8i z8 = {0, 0, 0, 0, 0, 0, 0, 0};
  __builtin_amdgcn_tensor_load_to_lds(g0, g1, z4, z4, z8, 0);
#else
  __builtin_amdgcn_tensor_load_to_lds(g0, g1, z4, z4, 0);
#endif
}

// ---------------- conversion ----------------

__global__ void f32_to_f16_kernel(const float* __restrict__ src,
                                  HALF* __restrict__ dst, int n) {
  int i = blockIdx.x * blockDim.x + threadIdx.x;
  if (i < n) dst[i] = (HALF)src[i];
}

// ---------------- GEMM:  Y[M,N] = X[M,K] * W[N,K]^T + bias, K=N=1024 ----------------
// 128 threads (4 waves), block tile 128(M) x 64(N).
// Wave w owns rows {w*16..w*16+15} and {64+w*16..64+w*16+15}: two A frags share
// the same 4 B frags -> 8 WMMAs per buffer, halving B traffic per FLOP.
// Explicit ping-pong fragment buffers: loads land in the registers the next
// WMMA consumes (no v_mov rotation), waits stay partial (in-order loads).

__global__ void __launch_bounds__(128)
gemm_xwT_kernel(const HALF* __restrict__ X, const HALF* __restrict__ W,
                const float* __restrict__ bias, float scale,
                HALF* __restrict__ outH, float* __restrict__ outF, int heads) {
  const int tid = threadIdx.x;
  const int l   = tid & 31;
  const int w   = tid >> 5;
  const int lm  = l & 15;
  const int hi8 = (l & 16) ? 8 : 0;
  const int n0  = blockIdx.x * 64;
  const int m0  = blockIdx.y * 128;

  v8f c[2][4];
#pragma unroll
  for (int g = 0; g < 2; ++g)
#pragma unroll
    for (int t = 0; t < 4; ++t)
#pragma unroll
      for (int r = 0; r < 8; ++r) c[g][t][r] = 0.0f;

  const HALF* xrow0 = X + (long long)(m0 + w * 16 + lm) * 1024;
  const HALF* xrow1 = X + (long long)(m0 + 64 + w * 16 + lm) * 1024;
  const HALF* wrow[4];
#pragma unroll
  for (int t = 0; t < 4; ++t)
    wrow[t] = W + (long long)(n0 + t * 16 + lm) * 1024;

  // ping-pong buffers: A at k, B at k+32
  v16h aA[2], aB[2], bA[4], bB[4];
  aA[0] = gather_a(xrow0, l);
  aA[1] = gather_a(xrow1, l);
#pragma unroll
  for (int t = 0; t < 4; ++t) bA[t] = gather_b(wrow[t], l);
  aB[0] = gather_a(xrow0 + 32, l);
  aB[1] = gather_a(xrow1 + 32, l);
#pragma unroll
  for (int t = 0; t < 4; ++t) bB[t] = gather_b(wrow[t] + 32, l);

  for (int k0 = 0; k0 < 1024; k0 += 64) {
    // ---- compute buffer A (k0); reload A for k0+64 underneath buffer-B math
#pragma unroll
    for (int t = 0; t < 4; ++t) c[0][t] = wmma_f16(aA[0], bA[t], c[0][t]);
#pragma unroll
    for (int t = 0; t < 4; ++t) c[1][t] = wmma_f16(aA[1], bA[t], c[1][t]);
    if (k0 + 64 < 1024) {
      aA[0] = gather_a(xrow0 + k0 + 64, l);
      aA[1] = gather_a(xrow1 + k0 + 64, l);
#pragma unroll
      for (int t = 0; t < 4; ++t) bA[t] = gather_b(wrow[t] + k0 + 64, l);
    }
    // ---- compute buffer B (k0+32); reload B for k0+96
#pragma unroll
    for (int t = 0; t < 4; ++t) c[0][t] = wmma_f16(aB[0], bB[t], c[0][t]);
#pragma unroll
    for (int t = 0; t < 4; ++t) c[1][t] = wmma_f16(aB[1], bB[t], c[1][t]);
    if (k0 + 96 < 1024) {
      aB[0] = gather_a(xrow0 + k0 + 96, l);
      aB[1] = gather_a(xrow1 + k0 + 96, l);
#pragma unroll
      for (int t = 0; t < 4; ++t) bB[t] = gather_b(wrow[t] + k0 + 96, l);
    }
  }

#pragma unroll
  for (int g = 0; g < 2; ++g) {
#pragma unroll
    for (int t = 0; t < 4; ++t) {
      const int n   = n0 + t * 16 + lm;
      const float bv = bias[n];
#pragma unroll
      for (int r = 0; r < 8; ++r) {
        const int mrow = m0 + g * 64 + w * 16 + r + hi8;
        const float val = (c[g][t][r] + bv) * scale;
        if (heads) {
          const int batch = mrow >> 11;
          const int s     = mrow & 2047;
          const int hh    = n >> 6;
          const int d     = n & 63;
          outH[(((long long)(batch * 16 + hh) * 2048 + s) * 64) + d] = (HALF)val;
        } else {
          outF[(long long)mrow * 1024 + n] = val;
        }
      }
    }
  }
}

// ---------------- Flash attention (causal), HD=64, S=2048 ----------------
// grid: (S/64, NH, B); block: 128 threads (4 waves). Wave w owns 16 query rows.
// V tile staged into LDS by the Tensor Data Mover; K score fragments stream
// from global (contiguous in d); next K tile prefetched.

__global__ void __launch_bounds__(128)
attn_kernel(const HALF* __restrict__ Q, const HALF* __restrict__ K,
            const HALF* __restrict__ V, HALF* __restrict__ O) {
  __shared__ HALF Vs[64 * 64];    // [key][d] row-major tile (TDM destination)
  __shared__ HALF Pl[64 * 64];    // [w*16+m][key] probabilities (wave-private rows)

  const int tid   = threadIdx.x;
  const int l     = tid & 31;
  const int w     = tid >> 5;
  const int m     = l & 15;
  const int hi8   = (l & 16) ? 8 : 0;
  const int q0    = blockIdx.x * 64;
  const int h     = blockIdx.y;
  const int batch = blockIdx.z;
  const long long bh = (long long)(batch * 16 + h) * 2048;

  // Q fragments (softmax scale folded in at projection time)
  const int qrow = q0 + w * 16 + m;
  const HALF* qb = Q + (bh + qrow) * 64;
  const v16h aQ0 = gather_a(qb, l);        // d 0..31
  const v16h aQ1 = gather_a(qb + 32, l);   // d 32..63

  const unsigned int vs_lds = (unsigned int)(size_t)(&Vs[0]);

  v8f   acc[4];
  float mrun[8], lrun[8];
#pragma unroll
  for (int r = 0; r < 8; ++r) { mrun[r] = -1e30f; lrun[r] = 0.0f; }
#pragma unroll
  for (int t = 0; t < 4; ++t)
#pragma unroll
    for (int r = 0; r < 8; ++r) acc[t][r] = 0.0f;

  for (int kv0 = 0; kv0 <= q0 + 63; kv0 += 64) {
    __syncthreads();                        // all waves done reading previous Vs
    if (w == 0) {                           // one TDM op per block per kv-block
      tdm_load_tile_64x64_f16(vs_lds, V + (bh + kv0) * 64);
      __builtin_amdgcn_s_wait_tensorcnt(0);
    }
    // prefetch next kv-block's K rows while this block computes
    if (kv0 + 64 <= q0 + 63)
      __builtin_prefetch(K + (bh + kv0 + 64 + (tid & 63)) * 64, 0, 1);
    __syncthreads();                        // Vs visible to all waves

    // ---- scores: S = Q * K^T
    v8f sc[4];
#pragma unroll
    for (int t = 0; t < 4; ++t) {
      v8f z;
#pragma unroll
      for (int r = 0; r < 8; ++r) z[r] = 0.0f;
      const HALF* kb = K + (bh + kv0 + t * 16 + m) * 64;  // lane's key column
      z = wmma_f16(aQ0, gather_b(kb, l), z);
      z = wmma_f16(aQ1, gather_b(kb + 32, l), z);
      sc[t] = z;
    }

    // ---- causal mask
#pragma unroll
    for (int t = 0; t < 4; ++t) {
      const int key = kv0 + t * 16 + m;
#pragma unroll
      for (int r = 0; r < 8; ++r) {
        const int qg = q0 + w * 16 + r + hi8;
        if (key > qg) sc[t][r] = -1e30f;
      }
    }

    // ---- online softmax (row reductions stay inside each 16-lane half)
#pragma unroll
    for (int r = 0; r < 8; ++r) {
      float mx = fmaxf(fmaxf(sc[0][r], sc[1][r]), fmaxf(sc[2][r], sc[3][r]));
#pragma unroll
      for (int off = 8; off >= 1; off >>= 1)
        mx = fmaxf(mx, __shfl_xor(mx, off, 32));
      const float mnew = fmaxf(mrun[r], mx);
      const float corr = __expf(mrun[r] - mnew);
      float s = 0.0f;
#pragma unroll
      for (int t = 0; t < 4; ++t) {
        const float p = __expf(sc[t][r] - mnew);
        sc[t][r] = p;
        s += p;
      }
#pragma unroll
      for (int off = 8; off >= 1; off >>= 1)
        s += __shfl_xor(s, off, 32);
      lrun[r] = lrun[r] * corr + s;
      mrun[r] = mnew;
#pragma unroll
      for (int t = 0; t < 4; ++t) acc[t][r] *= corr;
    }

    // ---- reshape P: C-layout -> LDS row-major (wave-private rows, no barrier)
#pragma unroll
    for (int t = 0; t < 4; ++t)
#pragma unroll
      for (int r = 0; r < 8; ++r)
        Pl[(w * 16 + r + hi8) * 64 + t * 16 + m] = (HALF)sc[t][r];

    // ---- O += P * V   (B-frags: strided gather over keys from row-major Vs)
    const HALF* prow = &Pl[(w * 16 + m) * 64];
    const v16h aP0 = gather_a(prow, l);       // keys 0..31
    const v16h aP1 = gather_a(prow + 32, l);  // keys 32..63
#pragma unroll
    for (int td = 0; td < 4; ++td) {
      const HALF* vb0 = &Vs[td * 16 + m];             // keys 0..31, lane's d column
      const HALF* vb1 = &Vs[32 * 64 + td * 16 + m];   // keys 32..63
      acc[td] = wmma_f16(aP0, gather_b_strided(vb0, 64, l), acc[td]);
      acc[td] = wmma_f16(aP1, gather_b_strided(vb1, 64, l), acc[td]);
    }
  }

  // ---- normalize and store to [B,S,H] f16
#pragma unroll
  for (int r = 0; r < 8; ++r) {
    const float inv = 1.0f / lrun[r];
    const int qg = q0 + w * 16 + r + hi8;
    const long long orow = ((long long)batch * 2048 + qg) * 1024 + h * 64;
#pragma unroll
    for (int td = 0; td < 4; ++td)
      O[orow + td * 16 + m] = (HALF)(acc[td][r] * inv);
  }
}

// ---------------- launch ----------------

extern "C" void kernel_launch(void* const* d_in, const int* in_sizes, int n_in,
                              void* d_out, int out_size, void* d_ws, size_t ws_size,
                              hipStream_t stream) {
  (void)in_sizes; (void)n_in; (void)out_size; (void)ws_size;

  const float* x  = (const float*)d_in[0];
  const float* Wq = (const float*)d_in[1];
  const float* bq = (const float*)d_in[2];
  const float* Wk = (const float*)d_in[3];
  const float* bk = (const float*)d_in[4];
  const float* Wv = (const float*)d_in[5];
  const float* bv = (const float*)d_in[6];
  const float* Wo = (const float*)d_in[7];
  const float* bo = (const float*)d_in[8];

  const int M = 4096;           // B*S
  const int Hd = 1024;          // hidden
  HALF* ws  = (HALF*)d_ws;
  HALF* Xh  = ws;                               // M*Hd
  HALF* Wqh = Xh  + (size_t)M * Hd;
  HALF* Wkh = Wqh + (size_t)Hd * Hd;
  HALF* Wvh = Wkh + (size_t)Hd * Hd;
  HALF* Woh = Wvh + (size_t)Hd * Hd;
  HALF* Qh  = Woh + (size_t)Hd * Hd;            // [B,NH,S,HD]
  HALF* Kh  = Qh  + (size_t)M * Hd;
  HALF* Vh  = Kh  + (size_t)M * Hd;
  HALF* Ah  = Vh  + (size_t)M * Hd;             // attn out [B,S,H]

  {
    const int nX = M * Hd;
    f32_to_f16_kernel<<<(nX + 255) / 256, 256, 0, stream>>>(x, Xh, nX);
    const int nW = Hd * Hd;
    f32_to_f16_kernel<<<(nW + 255) / 256, 256, 0, stream>>>(Wq, Wqh, nW);
    f32_to_f16_kernel<<<(nW + 255) / 256, 256, 0, stream>>>(Wk, Wkh, nW);
    f32_to_f16_kernel<<<(nW + 255) / 256, 256, 0, stream>>>(Wv, Wvh, nW);
    f32_to_f16_kernel<<<(nW + 255) / 256, 256, 0, stream>>>(Wo, Woh, nW);
  }

  dim3 gg(Hd / 64, M / 128);    // (16, 32)
  dim3 bb(128);
  gemm_xwT_kernel<<<gg, bb, 0, stream>>>(Xh, Wqh, bq, 0.125f, Qh, nullptr, 1);
  gemm_xwT_kernel<<<gg, bb, 0, stream>>>(Xh, Wkh, bk, 1.0f,   Kh, nullptr, 1);
  gemm_xwT_kernel<<<gg, bb, 0, stream>>>(Xh, Wvh, bv, 1.0f,   Vh, nullptr, 1);

  dim3 ga(2048 / 64, 16, 2);
  attn_kernel<<<ga, bb, 0, stream>>>(Qh, Kh, Vh, Ah);

  gemm_xwT_kernel<<<gg, bb, 0, stream>>>(Ah, Woh, bo, 1.0f, nullptr, (float*)d_out, 0);
}
